// relation_model_6158983102973
// MI455X (gfx1250) — matrix operand order repaired and checked by reading, not compiled, not run on hardware
//
#include <hip/hip_runtime.h>
#include <hip/hip_bf16.h>

// ---------- types ----------
typedef __attribute__((ext_vector_type(16))) __bf16 bf16x16;
typedef __attribute__((ext_vector_type(8)))  float  f32x8;
typedef __attribute__((ext_vector_type(4)))  unsigned int u32x4;

union Frag16 { bf16x16 v; u32x4 u[2]; };

// ---------- ROI align:  out[sample][pos(3x3)][c(1024)] (bf16) ----------
__global__ __launch_bounds__(256)
void roi_align_kernel(const float* __restrict__ fm_base,   // [B][1024][14][14]
                      const float* __restrict__ boxes,     // [B][perB][4] or nullptr (global box)
                      __bf16* __restrict__ out,            // [nSamples][9][1024]
                      int perB)
{
    int sidx = blockIdx.x;
    int b = sidx / perB;
    int rb = sidx - b * perB;
    float x1, y1, x2, y2;
    if (boxes) {
        const float* bx = boxes + ((size_t)b * perB + rb) * 4;
        const float sc = 14.0f / 448.0f;
        x1 = bx[0] * sc; y1 = bx[1] * sc; x2 = bx[2] * sc; y2 = bx[3] * sc;
    } else { x1 = 0.f; y1 = 0.f; x2 = 14.f; y2 = 14.f; }
    float bw = fmaxf(x2 - x1, 1.0f) * (1.0f / 3.0f);
    float bh = fmaxf(y2 - y1, 1.0f) * (1.0f / 3.0f);

    int y0i[6], y1i[6], x0i[6], x1i[6];
    float ly[6], lx[6];
#pragma unroll
    for (int j = 0; j < 6; ++j) {
        float off = 0.25f + 0.5f * (float)(j & 1);
        float p = (float)(j >> 1) + off;
        float gy = y1 + p * bh;
        float gx = x1 + p * bw;
        gy = fminf(fmaxf(gy, 0.0f), 13.0f);
        gx = fminf(fmaxf(gx, 0.0f), 13.0f);
        int yy0 = (int)floorf(gy), xx0 = (int)floorf(gx);
        y0i[j] = yy0; x0i[j] = xx0;
        y1i[j] = yy0 + 1 > 13 ? 13 : yy0 + 1;
        x1i[j] = xx0 + 1 > 13 ? 13 : xx0 + 1;
        ly[j] = gy - (float)yy0; lx[j] = gx - (float)xx0;
    }
    const float* fm = fm_base + (size_t)b * 1024 * 196;
    __bf16* o = out + (size_t)sidx * 9 * 1024;
    for (int c = threadIdx.x; c < 1024; c += 256) {
        const float* f = fm + (size_t)c * 196;
#pragma unroll
        for (int oy = 0; oy < 3; ++oy)
#pragma unroll
        for (int ox = 0; ox < 3; ++ox) {
            float acc = 0.f;
#pragma unroll
            for (int i = 0; i < 2; ++i)
#pragma unroll
            for (int j = 0; j < 2; ++j) {
                int yi = oy * 2 + i, xi = ox * 2 + j;
                float wy = ly[yi], wx = lx[xi];
                float v00 = f[y0i[yi] * 14 + x0i[xi]];
                float v01 = f[y0i[yi] * 14 + x1i[xi]];
                float v10 = f[y1i[yi] * 14 + x0i[xi]];
                float v11 = f[y1i[yi] * 14 + x1i[xi]];
                acc += v00 * (1.f - wy) * (1.f - wx) + v01 * (1.f - wy) * wx
                     + v10 * wy * (1.f - wx) + v11 * wy * wx;
            }
            o[(oy * 3 + ox) * 1024 + c] = (__bf16)(acc * 0.25f);
        }
    }
}

// ---------- BN fold: scale = g*rsqrt(v+eps), bias = b - m*scale ----------
__global__ void bn_prep_kernel(const float* __restrict__ g, const float* __restrict__ b,
                               const float* __restrict__ m, const float* __restrict__ v,
                               float* __restrict__ sc, float* __restrict__ bi, int n)
{
    int i = blockIdx.x * blockDim.x + threadIdx.x;
    if (i < n) {
        float inv = g[i] * rsqrtf(v[i] + 1e-5f);
        sc[i] = inv;
        bi[i] = b[i] - m[i] * inv;
    }
}

// ---------- weight repacks ----------
// conv W [co][ci][3][3] f32 -> Wg [co][(ky*3+kx)*Cin + ci] bf16
__global__ void repack_conv_w(const float* __restrict__ W, __bf16* __restrict__ Wg,
                              int Cout, int Cin)
{
    int idx = blockIdx.x * blockDim.x + threadIdx.x;
    int total = Cout * Cin * 9;
    if (idx >= total) return;
    int t = idx % 9;
    int rest = idx / 9;
    int ci = rest % Cin;
    int co = rest / Cin;
    Wg[(size_t)co * (9 * Cin) + t * Cin + ci] = (__bf16)W[idx];
}

// conv1d W [N][K][3], slice [:,:,1] -> bf16 [N][K]
__global__ void repack_c1d_w(const float* __restrict__ W, __bf16* __restrict__ Wg, int total)
{
    int idx = blockIdx.x * blockDim.x + threadIdx.x;
    if (idx < total) Wg[idx] = (__bf16)W[(size_t)idx * 3 + 1];
}

__global__ void f32_to_bf16_kernel(const float* __restrict__ src, __bf16* __restrict__ dst, int total)
{
    int i = blockIdx.x * blockDim.x + threadIdx.x;
    if (i < total) dst[i] = (__bf16)src[i];
}

__global__ void zero_bf16_kernel(__bf16* __restrict__ dst, int total)
{
    int i = blockIdx.x * blockDim.x + threadIdx.x;
    if (i < total) dst[i] = (__bf16)0.0f;
}

// fc.W2 [13][1024] -> [64][1024] bf16 zero-padded; b2 -> b2p[64]
__global__ void pad_fc2_kernel(const float* __restrict__ W2, const float* __restrict__ b2,
                               __bf16* __restrict__ Wg, float* __restrict__ b2p)
{
    int idx = blockIdx.x * blockDim.x + threadIdx.x;
    if (idx >= 64 * 1024) return;
    int n = idx >> 10, k = idx & 1023;
    Wg[idx] = (n < 13) ? (__bf16)W2[n * 1024 + k] : (__bf16)0.0f;
    if (idx < 64) b2p[idx] = (idx < 13) ? b2[idx] : 0.0f;
}

// ---------- WMMA GEMM: C[M][Nstore] = act( A(MxK) * W(NxK)^T * scale + bias ) ----------
// Block tile 128 x (NT*16), 8 wave32 waves; each wave a 16 x (NT*16) strip, K step 32.
// IM2COL: A is X[sample][9][Cin] bf16; halo taps read a zero page (no exec masking),
// and the 3x3-tap loop is hoisted so the inner loop is pure load+WMMA.
template <bool IM2COL, int NT>
__global__ __launch_bounds__(256)
void gemm_bf16_kernel(const __bf16* __restrict__ A,
                      const __bf16* __restrict__ W,
                      const __bf16* __restrict__ zpage,  // >= Cin zeros (im2col only)
                      float* __restrict__ Cf, __bf16* __restrict__ Ch,
                      const float* __restrict__ scale, const float* __restrict__ bias,
                      int K, int Nstore, int relu, int cinShift)
{
    const int lane = threadIdx.x & 31;
    const int wave = threadIdx.x >> 5;
    const int m0 = blockIdx.y * 128 + wave * 16;
    const int n0 = blockIdx.x * (NT * 16);
    const int half = lane >> 4;
    const int l15 = lane & 15;

    f32x8 acc[NT];
    const f32x8 fz = {0, 0, 0, 0, 0, 0, 0, 0};
#pragma unroll
    for (int t = 0; t < NT; ++t) acc[t] = fz;

    // B-fragment: lane holds column n, K groups [k+half*8,+8) and [k+16+half*8,+8)
    const __bf16* wp[NT];
#pragma unroll
    for (int t = 0; t < NT; ++t)
        wp[t] = W + (size_t)(n0 + t * 16 + l15) * K + half * 8;

    // A-fragment row owned by this lane (16-bit A layout: lanes 0-15 / 16-31 same M,
    // K groups [k+half*8,+8) and [k+16+half*8,+8))
    const int arow = m0 + l15;

    if (IM2COL) {
        const int Cin = 1 << cinShift;
        int s = arow / 9;
        int pos = arow - s * 9;
        int oy = pos / 3, ox = pos - oy * 3;
        const __bf16* Xs = A + ((size_t)s * 9 << cinShift);
        for (int tap = 0; tap < 9; ++tap) {
            int ky = tap / 3, kx = tap - ky * 3;
            int iy = oy + ky - 1, ix = ox + kx - 1;
            bool valid = ((unsigned)iy < 3u) & ((unsigned)ix < 3u);
            const __bf16* xb = valid ? (Xs + ((size_t)(iy * 3 + ix) << cinShift)) : zpage;
            const __bf16* ap = xb + half * 8;
            const int kb = tap << cinShift;
            for (int kc = 0; kc < Cin; kc += 32) {
                Frag16 a;
                a.u[0] = *(const u32x4*)(ap + kc);
                a.u[1] = *(const u32x4*)(ap + kc + 16);
                const int kk = kb + kc;
#pragma unroll
                for (int t = 0; t < NT; ++t) {
                    Frag16 b;
                    b.u[0] = *(const u32x4*)(wp[t] + kk);
                    b.u[1] = *(const u32x4*)(wp[t] + kk + 16);
                    acc[t] = __builtin_amdgcn_wmma_f32_16x16x32_bf16(
                        false, a.v, false, b.v, (short)0, acc[t], false, false);
                }
            }
        }
    } else {
        const __bf16* ap = A + (size_t)arow * K + half * 8;
        for (int k = 0; k < K; k += 32) {
            Frag16 a;
            a.u[0] = *(const u32x4*)(ap + k);
            a.u[1] = *(const u32x4*)(ap + k + 16);
            if (k + 1024 < K) __builtin_prefetch(ap + k + 1024, 0, 0);
#pragma unroll
            for (int t = 0; t < NT; ++t) {
                Frag16 b;
                b.u[0] = *(const u32x4*)(wp[t] + k);
                b.u[1] = *(const u32x4*)(wp[t] + k + 16);
                acc[t] = __builtin_amdgcn_wmma_f32_16x16x32_bf16(
                    false, a.v, false, b.v, (short)0, acc[t], false, false);
            }
        }
    }

    // C layout: element r of acc -> row m0 + half*8 + r, col = tile_n + l15
    const int rbase = m0 + half * 8;
#pragma unroll
    for (int t = 0; t < NT; ++t) {
        int n = n0 + t * 16 + l15;
        float scv = scale ? scale[n] : 1.0f;
        float biv = bias ? bias[n] : 0.0f;
#pragma unroll
        for (int r = 0; r < 8; ++r) {
            float vv = acc[t][r] * scv + biv;
            if (relu) vv = fmaxf(vv, 0.0f);
            if (n < Nstore) {
                size_t off = (size_t)(rbase + r) * Nstore + n;
                if (Cf) Cf[off] = vv; else Ch[off] = (__bf16)vv;
            }
        }
    }
}

// ---------- glue ----------
// pin[b][c*9+pos] = feat_p[(128 + b*8)][pos][c] + feat_p[b][pos][c]   (g_feat add)
__global__ void combine_pg_kernel(const float* __restrict__ feat_p, __bf16* __restrict__ pin)
{
    int idx = blockIdx.x * blockDim.x + threadIdx.x;
    if (idx >= 128 * 2304) return;
    int b = idx / 2304;
    int q = idx - b * 2304;
    int c = q / 9, pos = q - c * 9;
    float v = feat_p[((size_t)(128 + b * 8) * 9 + pos) * 256 + c]
            + feat_p[((size_t)b * 9 + pos) * 256 + c];
    pin[idx] = (__bf16)v;
}

// oin[s][c*9+pos] = feat_o[s][pos][c]
__global__ void transpose_o_kernel(const __bf16* __restrict__ feat_o, __bf16* __restrict__ oin)
{
    int idx = blockIdx.x * blockDim.x + threadIdx.x;
    if (idx >= 1024 * 2304) return;
    int sidx = idx / 2304;
    int q = idx - sidx * 2304;
    int c = q / 9, pos = q - c * 9;
    oin[idx] = feat_o[((size_t)sidx * 9 + pos) * 256 + c];
}

// rf[s][k] = k<2304 ? pf[s/8][k] : of[s][k-2304]
__global__ void concat_rf_kernel(const __bf16* __restrict__ pf, const __bf16* __restrict__ of_,
                                 __bf16* __restrict__ rf)
{
    int idx = blockIdx.x * blockDim.x + threadIdx.x;
    if (idx >= 1024 * 4608) return;
    int sidx = idx / 4608;
    int k = idx - sidx * 4608;
    rf[idx] = (k < 2304) ? pf[(size_t)(sidx >> 3) * 2304 + k]
                         : of_[(size_t)sidx * 2304 + (k - 2304)];
}

// ---------- host ----------
static inline int cdiv(int a, int b) { return (a + b - 1) / b; }

extern "C" void kernel_launch(void* const* d_in, const int* in_sizes, int n_in,
                              void* d_out, int out_size, void* d_ws, size_t ws_size,
                              hipStream_t stream)
{
    const float* fmap   = (const float*)d_in[0];
    const float* objf   = (const float*)d_in[1];
    const float* pboxes = (const float*)d_in[2];
    const float* oboxes = (const float*)d_in[3];
    const float* pW1 = (const float*)d_in[4];
    const float* pg1 = (const float*)d_in[5];
    const float* pb1 = (const float*)d_in[6];
    const float* pm1 = (const float*)d_in[7];
    const float* pv1 = (const float*)d_in[8];
    const float* pW2 = (const float*)d_in[9];
    const float* pg2 = (const float*)d_in[10];
    const float* pb2 = (const float*)d_in[11];
    const float* pm2 = (const float*)d_in[12];
    const float* pv2 = (const float*)d_in[13];
    const float* oW1 = (const float*)d_in[14];
    const float* og1 = (const float*)d_in[15];
    const float* ob1 = (const float*)d_in[16];
    const float* om1 = (const float*)d_in[17];
    const float* ov1 = (const float*)d_in[18];
    const float* oW2 = (const float*)d_in[19];
    const float* og2 = (const float*)d_in[20];
    const float* ob2 = (const float*)d_in[21];
    const float* om2 = (const float*)d_in[22];
    const float* ov2 = (const float*)d_in[23];
    const float* cpW1 = (const float*)d_in[24];
    const float* cpb1 = (const float*)d_in[25];
    const float* cpW2 = (const float*)d_in[26];
    const float* cpb2 = (const float*)d_in[27];
    const float* coW1 = (const float*)d_in[28];
    const float* cob1 = (const float*)d_in[29];
    const float* coW2 = (const float*)d_in[30];
    const float* cob2 = (const float*)d_in[31];
    const float* fW1 = (const float*)d_in[32];
    const float* fb1 = (const float*)d_in[33];
    const float* fW2 = (const float*)d_in[34];
    const float* fb2 = (const float*)d_in[35];

    char* ws = (char*)d_ws;
    size_t cur = 0;
    auto alloc = [&](size_t bytes) -> void* {
        void* p = ws + cur;
        cur = (cur + bytes + 255) & ~(size_t)255;
        return p;
    };

    // bf16 weights
    __bf16* wg_p1 = (__bf16*)alloc((size_t)512 * 9216 * 2);
    __bf16* wg_p2 = (__bf16*)alloc((size_t)256 * 4608 * 2);
    __bf16* wg_o1 = (__bf16*)alloc((size_t)512 * 9216 * 2);
    __bf16* wg_o2 = (__bf16*)alloc((size_t)256 * 4608 * 2);
    __bf16* wc_p1 = (__bf16*)alloc((size_t)2304 * 2304 * 2);
    __bf16* wc_p2 = (__bf16*)alloc((size_t)2304 * 2304 * 2);
    __bf16* wc_o1 = (__bf16*)alloc((size_t)2304 * 2304 * 2);
    __bf16* wc_o2 = (__bf16*)alloc((size_t)2304 * 2304 * 2);
    __bf16* wfc1  = (__bf16*)alloc((size_t)1024 * 4608 * 2);
    __bf16* wfc2  = (__bf16*)alloc((size_t)64 * 1024 * 2);
    float* b2p    = (float*)alloc(64 * 4);
    float* sc_p1 = (float*)alloc(512 * 4); float* bi_p1 = (float*)alloc(512 * 4);
    float* sc_p2 = (float*)alloc(256 * 4); float* bi_p2 = (float*)alloc(256 * 4);
    float* sc_o1 = (float*)alloc(512 * 4); float* bi_o1 = (float*)alloc(512 * 4);
    float* sc_o2 = (float*)alloc(256 * 4); float* bi_o2 = (float*)alloc(256 * 4);
    __bf16* zpage = (__bf16*)alloc((size_t)1024 * 2);   // zero page for halo taps

    // activations
    __bf16* roi_pg = (__bf16*)alloc((size_t)1152 * 9 * 1024 * 2); // [0,128)=global, [128,1152)=person
    __bf16* roi_o  = (__bf16*)alloc((size_t)1024 * 9 * 1024 * 2);
    __bf16* h1_p   = (__bf16*)alloc((size_t)1152 * 9 * 512 * 2);
    __bf16* h1_o   = (__bf16*)alloc((size_t)1024 * 9 * 512 * 2);
    float*  feat_p = (float*) alloc((size_t)1152 * 9 * 256 * 4);
    __bf16* feat_o = (__bf16*)alloc((size_t)1024 * 9 * 256 * 2);
    __bf16* pin    = (__bf16*)alloc((size_t)128 * 2304 * 2);
    __bf16* oin    = (__bf16*)alloc((size_t)1024 * 2304 * 2);
    __bf16* pf_h   = (__bf16*)alloc((size_t)128 * 2304 * 2);
    __bf16* pf     = (__bf16*)alloc((size_t)128 * 2304 * 2);
    __bf16* of_h   = (__bf16*)alloc((size_t)1024 * 2304 * 2);
    __bf16* of_b   = (__bf16*)alloc((size_t)1024 * 2304 * 2);
    __bf16* rf     = (__bf16*)alloc((size_t)1024 * 4608 * 2);
    __bf16* fc_h   = (__bf16*)alloc((size_t)1024 * 1024 * 2);
    (void)ws_size; (void)in_sizes; (void)n_in; (void)out_size;

    // --- weight prep ---
    bn_prep_kernel<<<cdiv(512, 256), 256, 0, stream>>>(pg1, pb1, pm1, pv1, sc_p1, bi_p1, 512);
    bn_prep_kernel<<<cdiv(256, 256), 256, 0, stream>>>(pg2, pb2, pm2, pv2, sc_p2, bi_p2, 256);
    bn_prep_kernel<<<cdiv(512, 256), 256, 0, stream>>>(og1, ob1, om1, ov1, sc_o1, bi_o1, 512);
    bn_prep_kernel<<<cdiv(256, 256), 256, 0, stream>>>(og2, ob2, om2, ov2, sc_o2, bi_o2, 256);
    zero_bf16_kernel<<<cdiv(1024, 256), 256, 0, stream>>>(zpage, 1024);

    repack_conv_w<<<cdiv(512 * 1024 * 9, 256), 256, 0, stream>>>(pW1, wg_p1, 512, 1024);
    repack_conv_w<<<cdiv(256 * 512 * 9, 256), 256, 0, stream>>>(pW2, wg_p2, 256, 512);
    repack_conv_w<<<cdiv(512 * 1024 * 9, 256), 256, 0, stream>>>(oW1, wg_o1, 512, 1024);
    repack_conv_w<<<cdiv(256 * 512 * 9, 256), 256, 0, stream>>>(oW2, wg_o2, 256, 512);

    const int c1dtot = 2304 * 2304;
    repack_c1d_w<<<cdiv(c1dtot, 256), 256, 0, stream>>>(cpW1, wc_p1, c1dtot);
    repack_c1d_w<<<cdiv(c1dtot, 256), 256, 0, stream>>>(cpW2, wc_p2, c1dtot);
    repack_c1d_w<<<cdiv(c1dtot, 256), 256, 0, stream>>>(coW1, wc_o1, c1dtot);
    repack_c1d_w<<<cdiv(c1dtot, 256), 256, 0, stream>>>(coW2, wc_o2, c1dtot);

    f32_to_bf16_kernel<<<cdiv(1024 * 4608, 256), 256, 0, stream>>>(fW1, wfc1, 1024 * 4608);
    pad_fc2_kernel<<<cdiv(64 * 1024, 256), 256, 0, stream>>>(fW2, fb2, wfc2, b2p);

    // --- ROI align: global (fixed box), person, object ---
    roi_align_kernel<<<128, 256, 0, stream>>>(fmap, nullptr, roi_pg, 1);
    roi_align_kernel<<<1024, 256, 0, stream>>>(fmap, pboxes, roi_pg + (size_t)128 * 9 * 1024, 8);
    roi_align_kernel<<<1024, 256, 0, stream>>>(objf, oboxes, roi_o, 8);

    // --- conv stacks as implicit-im2col WMMA GEMMs ---
    // person+global: M=1152*9=10368
    gemm_bf16_kernel<true, 8><<<dim3(512 / 128, 10368 / 128), 256, 0, stream>>>(
        roi_pg, wg_p1, zpage, nullptr, h1_p, sc_p1, bi_p1, 9216, 512, 1, 10);
    gemm_bf16_kernel<true, 8><<<dim3(256 / 128, 10368 / 128), 256, 0, stream>>>(
        h1_p, wg_p2, zpage, feat_p, nullptr, sc_p2, bi_p2, 4608, 256, 1, 9);
    // object: M=1024*9=9216
    gemm_bf16_kernel<true, 8><<<dim3(512 / 128, 9216 / 128), 256, 0, stream>>>(
        roi_o, wg_o1, zpage, nullptr, h1_o, sc_o1, bi_o1, 9216, 512, 1, 10);
    gemm_bf16_kernel<true, 8><<<dim3(256 / 128, 9216 / 128), 256, 0, stream>>>(
        h1_o, wg_o2, zpage, nullptr, feat_o, sc_o2, bi_o2, 4608, 256, 1, 9);

    // --- feature vectors (add g_feat, transpose to channel-major 2304) ---
    combine_pg_kernel<<<cdiv(128 * 2304, 256), 256, 0, stream>>>(feat_p, pin);
    transpose_o_kernel<<<cdiv(1024 * 2304, 256), 256, 0, stream>>>(feat_o, oin);

    // --- conv1d_len1 = two dense FCs each ---
    gemm_bf16_kernel<false, 8><<<dim3(2304 / 128, 128 / 128), 256, 0, stream>>>(
        pin, wc_p1, nullptr, nullptr, pf_h, nullptr, cpb1, 2304, 2304, 1, 0);
    gemm_bf16_kernel<false, 8><<<dim3(2304 / 128, 128 / 128), 256, 0, stream>>>(
        pf_h, wc_p2, nullptr, nullptr, pf, nullptr, cpb2, 2304, 2304, 1, 0);
    gemm_bf16_kernel<false, 8><<<dim3(2304 / 128, 1024 / 128), 256, 0, stream>>>(
        oin, wc_o1, nullptr, nullptr, of_h, nullptr, cob1, 2304, 2304, 1, 0);
    gemm_bf16_kernel<false, 8><<<dim3(2304 / 128, 1024 / 128), 256, 0, stream>>>(
        of_h, wc_o2, nullptr, nullptr, of_b, nullptr, cob2, 2304, 2304, 1, 0);

    // --- concat + final FC ---
    concat_rf_kernel<<<cdiv(1024 * 4608, 256), 256, 0, stream>>>(pf, of_b, rf);
    gemm_bf16_kernel<false, 8><<<dim3(1024 / 128, 1024 / 128), 256, 0, stream>>>(
        rf, wfc1, nullptr, nullptr, fc_h, nullptr, fb1, 4608, 1024, 1, 0);
    gemm_bf16_kernel<false, 4><<<dim3(1, 1024 / 128), 256, 0, stream>>>(
        fc_h, wfc2, nullptr, (float*)d_out, nullptr, nullptr, b2p, 1024, 13, 0, 0);
}
